// TriangleAttention_43044162241204
// MI455X (gfx1250) — matrix operand order repaired and checked by reading, not compiled, not run on hardware
//
#include <hip/hip_runtime.h>
#include <hip/hip_fp16.h>

typedef __attribute__((ext_vector_type(16))) _Float16 v16h;
typedef __attribute__((ext_vector_type(8)))  _Float16 v8h;
typedef __attribute__((ext_vector_type(8)))  float    v8f;
typedef __attribute__((ext_vector_type(2)))  int      v2i;

#define AS1 __attribute__((address_space(1)))
#define AS3 __attribute__((address_space(3)))

#define N_SEQ   384
#define CZ      128
#define N_HEADS 4
#define D_HEAD  32
#define NTOK    (N_SEQ * N_SEQ)      // 147456
#define QTILES  (N_SEQ / 16)         // 24
#define KCHUNKS (N_SEQ / 32)         // 12
#define QK_SCALE 0.17677669529663689f
#define RMS_EPS  1e-5f
#define MASK_NEG (-30000.0f)         // f16-safe "-inf"

// Async global->LDS (CDNA5): use when toolchain declares the builtins.
#if defined(__has_builtin)
#if __has_builtin(__builtin_amdgcn_global_load_async_to_lds_b64) && \
    __has_builtin(__builtin_amdgcn_s_wait_asynccnt)
#define USE_ASYNC_LDS 1
#endif
#endif
#ifndef USE_ASYNC_LDS
#define USE_ASYNC_LDS 0
#endif

// ---------------------------------------------------------------------------
// WMMA helpers (layouts per CDNA5 ISA §7.12.2, wave32)
// ---------------------------------------------------------------------------
__device__ __forceinline__ v8f wmma_fp16(v16h a, v16h b, v8f c) {
  return __builtin_amdgcn_wmma_f32_16x16x32_f16(false, a, false, b,
                                                (short)0, c, false, false);
}

__device__ __forceinline__ v16h pack16(v8h lo, v8h hi) {
  v16h r;
#pragma unroll
  for (int i = 0; i < 8; ++i) { r[i] = lo[i]; r[i + 8] = hi[i]; }
  return r;
}

// A fragment: 16x32 f16 tile from row-major [16 x ld] memory, K window at kbase.
__device__ __forceinline__ v16h load_a_frag(const _Float16* base, int ld,
                                            int kbase, int lane) {
  const _Float16* p = base + (size_t)(lane & 15) * ld + kbase + ((lane >> 4) << 3);
  return pack16(*(const v8h*)p, *(const v8h*)(p + 16));
}

// B fragment: element (k, n) at baseT[(nbase+n)*ld + kbase + k]  (memory [N][K])
__device__ __forceinline__ v16h load_b_frag(const _Float16* baseT, int ld,
                                            int nbase, int kbase, int lane) {
  const _Float16* p = baseT + (size_t)(nbase + (lane & 15)) * ld + kbase
                    + ((lane >> 4) << 3);
  return pack16(*(const v8h*)p, *(const v8h*)(p + 16));
}

// 8-byte global -> LDS copy (async when available)
__device__ __forceinline__ void copy8_g2l(const _Float16* src, _Float16* dst) {
#if USE_ASYNC_LDS
  __builtin_amdgcn_global_load_async_to_lds_b64(
      (AS1 v2i*)(unsigned long long)(const void*)src,
      (AS3 v2i*)dst, 0, 0);
#else
  *(unsigned long long*)dst = *(const unsigned long long*)src;
#endif
}

__device__ __forceinline__ void wait_stage_done() {
#if USE_ASYNC_LDS
  __builtin_amdgcn_s_wait_asynccnt(0);
#endif
  __syncthreads();
}

// ---------------------------------------------------------------------------
// k0: transpose 5 weight matrices [in][out] f32 -> [out][in] f16 (B-frag ready)
// ---------------------------------------------------------------------------
__global__ void k0_weight_cvt(const float* __restrict__ wq, const float* __restrict__ wk,
                              const float* __restrict__ wv, const float* __restrict__ wg,
                              const float* __restrict__ wo, _Float16* __restrict__ dst) {
  int idx = blockIdx.x * blockDim.x + threadIdx.x;
  if (idx >= 5 * CZ * CZ) return;
  int m = idx >> 14;
  int r = idx & 16383;
  int i = r >> 7;       // input channel
  int o = r & 127;      // output channel
  const float* src = (m == 0) ? wq : (m == 1) ? wk : (m == 2) ? wv
                   : (m == 3) ? wg : wo;
  dst[(size_t)m * (CZ * CZ) + o * CZ + i] = (_Float16)src[i * CZ + o];
}

// ---------------------------------------------------------------------------
// k1: RMSNorm + Q/K/V/G projections (WMMA) + pair bias + mask bias
// ---------------------------------------------------------------------------
__global__ __launch_bounds__(256)
void k1_norm_proj(const float* __restrict__ z, const unsigned char* __restrict__ zmask,
                  const float* __restrict__ norm_w,
                  const _Float16* __restrict__ wqT, const _Float16* __restrict__ wkT,
                  const _Float16* __restrict__ wvT, const _Float16* __restrict__ wgT,
                  const float* __restrict__ wz, const float* __restrict__ bg,
                  _Float16* __restrict__ Q, _Float16* __restrict__ K,
                  _Float16* __restrict__ Vt, _Float16* __restrict__ G,
                  _Float16* __restrict__ BIAS) {
  __shared__ _Float16 zn[8][16][136];
  const int lane = threadIdx.x & 31;
  const int wv_  = threadIdx.x >> 5;
  const int tile = blockIdx.x * 8 + wv_;
  const int t0   = tile * 16;
  const int tok  = lane & 15;
  const int half = lane >> 4;
  const int t    = t0 + tok;

  // --- RMSNorm: 2 lanes per token, 64 channels each ---
  const float* zp = z + (size_t)t * CZ + half * 64;
  float vals[64];
  float ss = 0.f;
#pragma unroll
  for (int j = 0; j < 16; ++j) {
    float4 v4 = ((const float4*)zp)[j];
    vals[4 * j + 0] = v4.x; vals[4 * j + 1] = v4.y;
    vals[4 * j + 2] = v4.z; vals[4 * j + 3] = v4.w;
    ss += v4.x * v4.x + v4.y * v4.y + v4.z * v4.z + v4.w * v4.w;
  }
  ss += __shfl_xor(ss, 16, 32);
  float rstd = rsqrtf(ss * (1.0f / (float)CZ) + RMS_EPS);
#pragma unroll
  for (int j = 0; j < 64; ++j) {
    float nw = norm_w[half * 64 + j];
    zn[wv_][tok][half * 64 + j] = (_Float16)(vals[j] * rstd * nw);
  }
  __syncthreads();

  // --- pair bias: bias[h, t] = zn . wz[:,h]  (+ mask bias), stored f16 ---
  {
    float a0 = 0.f, a1 = 0.f;
    int h0 = half * 2;
#pragma unroll 8
    for (int c = 0; c < CZ; ++c) {
      float v = (float)zn[wv_][tok][c];
      a0 += v * wz[c * N_HEADS + h0];
      a1 += v * wz[c * N_HEADS + h0 + 1];
    }
    float mb = zmask[t] ? 0.f : MASK_NEG;
    BIAS[(size_t)h0 * NTOK + t]       = (_Float16)(a0 + mb);
    BIAS[(size_t)(h0 + 1) * NTOK + t] = (_Float16)(a1 + mb);
  }

  v16h a[4];
#pragma unroll
  for (int kt = 0; kt < 4; ++kt)
    a[kt] = load_a_frag(&zn[wv_][0][0], 136, kt * 32, lane);

  const int bb     = t0 / N_SEQ;
  const int sbase  = t0 % N_SEQ;
  const int n_lane = lane & 15;
  const int mbase  = (lane >> 4) * 8;

#pragma unroll 1
  for (int nt = 0; nt < 8; ++nt) {
    v8f accq = {}, acck = {}, accv = {}, accg = {};
#pragma unroll
    for (int kt = 0; kt < 4; ++kt) {
      accq = wmma_fp16(a[kt], load_b_frag(wqT, CZ, nt * 16, kt * 32, lane), accq);
      acck = wmma_fp16(a[kt], load_b_frag(wkT, CZ, nt * 16, kt * 32, lane), acck);
      accv = wmma_fp16(a[kt], load_b_frag(wvT, CZ, nt * 16, kt * 32, lane), accv);
      accg = wmma_fp16(a[kt], load_b_frag(wgT, CZ, nt * 16, kt * 32, lane), accg);
    }
    int ng = nt * 16 + n_lane;
    int h  = ng >> 5, d = ng & 31;
    float bgv = bg[ng];
#pragma unroll
    for (int r = 0; r < 8; ++r) {
      int m = mbase + r;
      int s = sbase + m;
      size_t qk = ((size_t)(bb * N_HEADS + h) * N_SEQ + s) * D_HEAD + d;
      Q[qk] = (_Float16)(accq[r] * QK_SCALE);
      K[qk] = (_Float16)acck[r];
      Vt[((size_t)(bb * N_HEADS + h) * D_HEAD + d) * N_SEQ + s] = (_Float16)accv[r];
      G[(size_t)(t0 + m) * CZ + ng] = (_Float16)(accg[r] + bgv);
    }
  }
}

// ---------------------------------------------------------------------------
// k2: flash attention. Block = 8 waves sharing (b, h), 8 consecutive q-tiles.
//     K/V 32-chunks double-buffered in LDS (async global->LDS when available).
// ---------------------------------------------------------------------------
__global__ __launch_bounds__(256)
void k2_attention(const _Float16* __restrict__ Q, const _Float16* __restrict__ K,
                  const _Float16* __restrict__ Vt, const _Float16* __restrict__ BIAS,
                  _Float16* __restrict__ AO) {
  __shared__ _Float16 Kc[2][32][40];     // [seq-in-chunk][d], padded
  __shared__ _Float16 Vc[2][32][40];     // [d][seq-in-chunk], padded
  __shared__ _Float16 P[8][16][40];      // per-wave P staging

  const int tid  = threadIdx.x;
  const int lane = tid & 31;
  const int wv_  = tid >> 5;
  const int bid  = blockIdx.x;
  const int qblk = bid % 3;
  const int h    = (bid / 3) % N_HEADS;
  const int bb   = bid / (3 * N_HEADS);
  const int qt   = qblk * 8 + wv_;       // this wave's q-tile (0..23)

  const _Float16* Qbh = Q  + (size_t)(bb * N_HEADS + h) * N_SEQ * D_HEAD;
  const _Float16* Kbh = K  + (size_t)(bb * N_HEADS + h) * N_SEQ * D_HEAD;
  const _Float16* Vbh = Vt + (size_t)(bb * N_HEADS + h) * D_HEAD * N_SEQ;
  const _Float16* Bh  = BIAS + (size_t)h * NTOK;

  // per-thread staging slices (K chunk is one contiguous 2KB block; V is 32x64B rows)
  const int kci = tid * 4;                    // K: half index in chunk
  const int kr = kci >> 5, kc = kci & 31;
  const int vd = tid >> 3, vcol = (tid & 7) * 4;

  // prologue: stage chunk 0 into buffer 0
  copy8_g2l(Kbh + kci, &Kc[0][kr][kc]);
  copy8_g2l(Vbh + (size_t)vd * N_SEQ + vcol, &Vc[0][vd][vcol]);
  wait_stage_done();

  v16h aq = load_a_frag(Qbh + (size_t)qt * 16 * D_HEAD, D_HEAD, 0, lane);

  const int n_lane = lane & 15;
  const int mbase  = (lane >> 4) * 8;
  const int qrow0  = qt * 16 + mbase;

  float m_i[8], l_i[8];
  v8f o0 = {}, o1 = {};
#pragma unroll
  for (int r = 0; r < 8; ++r) { m_i[r] = -__builtin_inff(); l_i[r] = 0.f; }

  int buf = 0;
#pragma unroll 1
  for (int c = 0; c < KCHUNKS; ++c) {
    const int kb = c * 32;
    if (c + 1 < KCHUNKS) {               // prefetch next chunk into other buffer
      const int kb2 = kb + 32;
      copy8_g2l(Kbh + (size_t)kb2 * D_HEAD + kci, &Kc[buf ^ 1][kr][kc]);
      copy8_g2l(Vbh + (size_t)vd * N_SEQ + kb2 + vcol, &Vc[buf ^ 1][vd][vcol]);
    }

    // bias (f16 in memory) as WMMA C operand
    v8f c0, c1;
#pragma unroll
    for (int r = 0; r < 8; ++r) {
      const _Float16* bp = Bh + (size_t)(qrow0 + r) * N_SEQ + kb + n_lane;
      c0[r] = (float)bp[0];
      c1[r] = (float)bp[16];
    }
    v16h bk0 = load_b_frag(&Kc[buf][0][0], 40, 0,  0, lane);
    v16h bk1 = load_b_frag(&Kc[buf][0][0], 40, 16, 0, lane);
    v8f s0 = wmma_fp16(aq, bk0, c0);
    v8f s1 = wmma_fp16(aq, bk1, c1);

    // online softmax (row = (lane>>4)*8 + r; row reduce = 16-lane xor tree)
    float p0[8], p1[8];
#pragma unroll
    for (int r = 0; r < 8; ++r) {
      float rm = fmaxf(s0[r], s1[r]);
#pragma unroll
      for (int off = 1; off < 16; off <<= 1)
        rm = fmaxf(rm, __shfl_xor(rm, off, 32));
      float nm   = fmaxf(m_i[r], rm);
      float corr = __expf(m_i[r] - nm);
      m_i[r] = nm;
      p0[r] = __expf(s0[r] - nm);
      p1[r] = __expf(s1[r] - nm);
      float rs = p0[r] + p1[r];
#pragma unroll
      for (int off = 1; off < 16; off <<= 1)
        rs += __shfl_xor(rs, off, 32);
      l_i[r] = l_i[r] * corr + rs;
      o0[r] *= corr;
      o1[r] *= corr;
    }

    // P (16q x 32k) C-layout -> LDS -> A fragment (intra-wave DS order suffices)
#pragma unroll
    for (int r = 0; r < 8; ++r) {
      P[wv_][mbase + r][n_lane]      = (_Float16)p0[r];
      P[wv_][mbase + r][n_lane + 16] = (_Float16)p1[r];
    }
    v16h ap  = load_a_frag(&P[wv_][0][0], 40, 0, lane);
    v16h bv0 = load_b_frag(&Vc[buf][0][0], 40, 0,  0, lane);
    v16h bv1 = load_b_frag(&Vc[buf][0][0], 40, 16, 0, lane);
    o0 = wmma_fp16(ap, bv0, o0);
    o1 = wmma_fp16(ap, bv1, o1);

    // next chunk loaded + everyone done reading current buffers
    wait_stage_done();
    buf ^= 1;
  }

  const int tokbase = bb * N_SEQ + qt * 16;
#pragma unroll
  for (int r = 0; r < 8; ++r) {
    float inv = 1.0f / l_i[r];
    size_t row = (size_t)(tokbase + mbase + r) * CZ + h * D_HEAD;
    AO[row + n_lane]      = (_Float16)(o0[r] * inv);
    AO[row + 16 + n_lane] = (_Float16)(o1[r] * inv);
  }
}

// ---------------------------------------------------------------------------
// k3: output projection + bias + gating: out = (AO @ wo + bo) * g
// ---------------------------------------------------------------------------
__global__ __launch_bounds__(256)
void k3_out_proj(const _Float16* __restrict__ AO, const _Float16* __restrict__ woT,
                 const float* __restrict__ bo, const _Float16* __restrict__ G,
                 float* __restrict__ out) {
  const int lane = threadIdx.x & 31;
  const int wv_  = threadIdx.x >> 5;
  const int tile = blockIdx.x * 8 + wv_;
  const int t0   = tile * 16;

  v16h a[4];
#pragma unroll
  for (int kt = 0; kt < 4; ++kt)
    a[kt] = load_a_frag(AO + (size_t)t0 * CZ, CZ, kt * 32, lane);

  const int n_lane = lane & 15;
  const int mbase  = (lane >> 4) * 8;

#pragma unroll 1
  for (int nt = 0; nt < 8; ++nt) {
    v8f acc = {};
#pragma unroll
    for (int kt = 0; kt < 4; ++kt)
      acc = wmma_fp16(a[kt], load_b_frag(woT, CZ, nt * 16, kt * 32, lane), acc);
    int ng = nt * 16 + n_lane;
    float bov = bo[ng];
#pragma unroll
    for (int r = 0; r < 8; ++r) {
      size_t idx = (size_t)(t0 + mbase + r) * CZ + ng;
      out[idx] = (acc[r] + bov) * (float)G[idx];
    }
  }
}

// ---------------------------------------------------------------------------
extern "C" void kernel_launch(void* const* d_in, const int* in_sizes, int n_in,
                              void* d_out, int out_size, void* d_ws, size_t ws_size,
                              hipStream_t stream) {
  const float*         z      = (const float*)d_in[0];
  const unsigned char* zmask  = (const unsigned char*)d_in[1];
  const float*         norm_w = (const float*)d_in[2];
  const float*         wq     = (const float*)d_in[3];
  const float*         wk     = (const float*)d_in[4];
  const float*         wv     = (const float*)d_in[5];
  const float*         wz     = (const float*)d_in[6];
  const float*         wg     = (const float*)d_in[7];
  const float*         bg     = (const float*)d_in[8];
  const float*         wo     = (const float*)d_in[9];
  const float*         bo     = (const float*)d_in[10];
  float*               out    = (float*)d_out;

  char* ws = (char*)d_ws;
  size_t off = 0;
  auto take = [&](size_t bytes) {
    void* p = ws + off;
    off = (off + bytes + 255) & ~(size_t)255;
    return p;
  };
  _Float16* WT   = (_Float16*)take((size_t)5 * CZ * CZ * sizeof(_Float16));
  _Float16* Qb   = (_Float16*)take((size_t)NTOK * CZ * sizeof(_Float16));
  _Float16* Kb   = (_Float16*)take((size_t)NTOK * CZ * sizeof(_Float16));
  _Float16* Vtb  = (_Float16*)take((size_t)NTOK * CZ * sizeof(_Float16));
  _Float16* Gb   = (_Float16*)take((size_t)NTOK * CZ * sizeof(_Float16));
  _Float16* AOb  = (_Float16*)take((size_t)NTOK * CZ * sizeof(_Float16));
  _Float16* BIAS = (_Float16*)take((size_t)N_HEADS * NTOK * sizeof(_Float16));
  (void)ws_size; (void)in_sizes; (void)n_in; (void)out_size;

  const _Float16* wqT = WT;
  const _Float16* wkT = WT + 1 * CZ * CZ;
  const _Float16* wvT = WT + 2 * CZ * CZ;
  const _Float16* wgT = WT + 3 * CZ * CZ;
  const _Float16* woT = WT + 4 * CZ * CZ;

  k0_weight_cvt<<<(5 * CZ * CZ + 255) / 256, 256, 0, stream>>>(wq, wk, wv, wg, wo, WT);

  k1_norm_proj<<<(NTOK / 16) / 8, 256, 0, stream>>>(
      z, zmask, norm_w, wqT, wkT, wvT, wgT, wz, bg, Qb, Kb, Vtb, Gb, BIAS);

  // 384 b * 4 h * 3 q-blocks (8 q-tiles each) = 4608 blocks
  k2_attention<<<N_SEQ * N_HEADS * 3, 256, 0, stream>>>(Qb, Kb, Vtb, BIAS, AOb);

  k3_out_proj<<<(NTOK / 16) / 8, 256, 0, stream>>>(AOb, woT, bo, Gb, out);
}
